// Qwen2MLP_25486335934611
// MI455X (gfx1250) — compile-verified, matrix operand order checked by reading, not run on hardware
//
#include <hip/hip_runtime.h>
#include <math.h>

// ---------------------------------------------------------------------------
// Qwen2 MLP w/ int4-range fake-quant, Hadamard rotation, int8 WMMA GEMMs.
// Target: gfx1250 (CDNA5, wave32, V_WMMA_I32_16X16X64_IU8).
// Register-blocked: each wave owns a 32x32 (GEMM1, gate+up) or 32x64 (GEMM2)
// output region -> 8 WMMAs per 6 fragment loads per K=64 step.
// ---------------------------------------------------------------------------

#define HIDDEN 4096
#define INTER  16384
#define MTOK   4096            // 2 * 2048 token rows

typedef int v8i __attribute__((ext_vector_type(8)));

// ---------------------------------------------------------------------------
// Per-row symmetric quantization: scale = max(amax/7, 1e-8),
// q = clip(rint(x/scale), -8, 7) stored as int8; scale kept fp32.
// One 256-thread block per row.
// ---------------------------------------------------------------------------
__global__ void q2mlp_quant_rows(const float* __restrict__ in,
                                 signed char* __restrict__ q,
                                 float* __restrict__ scales,
                                 int C) {
    __shared__ float red[256];
    const int t = threadIdx.x;
    const size_t base = (size_t)blockIdx.x * (size_t)C;

    float m = 0.0f;
    for (int i = t; i < C; i += 256) m = fmaxf(m, fabsf(in[base + i]));
    red[t] = m;
    __syncthreads();
    for (int s = 128; s > 0; s >>= 1) {
        if (t < s) red[t] = fmaxf(red[t], red[t + s]);
        __syncthreads();
    }
    const float scale = fmaxf(red[0] * (1.0f / 7.0f), 1e-8f);
    const float inv = 1.0f / scale;

    for (int i = t; i < C; i += 256) {
        float v = rintf(in[base + i] * inv);         // RNE == jnp.round
        v = fminf(fmaxf(v, -8.0f), 7.0f);
        q[base + i] = (signed char)v;
    }
    if (t == 0) scales[blockIdx.x] = scale;
}

// ---------------------------------------------------------------------------
// Fragment loader for V_WMMA_I32_16X16X64_IU8.
// 8-bit A 16x64 layout (ISA 7.12.2): lane = m + 16*half, VGPR pair j holds
// 8 consecutive K bytes at k = half*8 + 16*j  (j = 0..3).  B columns are
// weight rows (row-major over K), so B uses the identical per-lane pattern.
// ---------------------------------------------------------------------------
__device__ __forceinline__ v8i q2mlp_load_frag(const signed char* p) {
    const int2* p2 = (const int2*)p;      // 8-byte chunks; stride 16B -> idx 0,2,4,6
    int2 c0 = p2[0];
    int2 c1 = p2[2];
    int2 c2 = p2[4];
    int2 c3 = p2[6];
    v8i f;
    f[0] = c0.x; f[1] = c0.y;
    f[2] = c1.x; f[3] = c1.y;
    f[4] = c2.x; f[5] = c2.y;
    f[6] = c3.x; f[7] = c3.y;
    return f;
}

__device__ __forceinline__ v8i q2mlp_wmma(v8i a, v8i b, v8i c) {
    return __builtin_amdgcn_wmma_i32_16x16x64_iu8(true, a, true, b, c,
                                                  false, false);
}

// ---------------------------------------------------------------------------
// GEMM1 + SiLU fusion:
//   h[m, o] = silu(sx[m]*sw[o]    * (xq[m,:] . wq[o,:]))
//           *      (sx[m]*sw[o+I] * (xq[m,:] . wq[o+I,:]))
// Block: 256 threads = 8 waves laid out 2 (M) x 4 (O); wave tile 32x32 for
// BOTH gate and up => 8 acc tiles, 8 WMMAs per 6 fragment loads per K-step.
// Grid: (INTER/128, MTOK/64).
// ---------------------------------------------------------------------------
__global__ void q2mlp_gemm1_silu(const signed char* __restrict__ xq,
                                 const float* __restrict__ xs,
                                 const signed char* __restrict__ wq,
                                 const float* __restrict__ wsc,
                                 float* __restrict__ h) {
    const int lane = threadIdx.x & 31;
    const int wave = threadIdx.x >> 5;
    const int wr   = wave >> 2;            // 0..1
    const int wc   = wave & 3;             // 0..3
    const int half = lane >> 4;            // 0..1
    const int l16  = lane & 15;

    const int m_base = blockIdx.y * 64 + wr * 32;
    const int o_base = blockIdx.x * 128 + wc * 32;

    const signed char* a0 = xq + (size_t)(m_base + l16) * HIDDEN + half * 8;
    const signed char* a1 = a0 + (size_t)16 * HIDDEN;
    const signed char* g0 = wq + (size_t)(o_base + l16) * HIDDEN + half * 8;
    const signed char* g1 = g0 + (size_t)16 * HIDDEN;
    const signed char* u0 = g0 + (size_t)INTER * HIDDEN;
    const signed char* u1 = g1 + (size_t)INTER * HIDDEN;

    v8i accg[2][2] = {{{}, {}}, {{}, {}}};
    v8i accu[2][2] = {{{}, {}}, {{}, {}}};

    for (int kk = 0; kk < HIDDEN; kk += 64) {
        __builtin_prefetch(g0 + kk + 256, 0, 0);   // global_prefetch_b8
        __builtin_prefetch(g1 + kk + 256, 0, 0);
        __builtin_prefetch(u0 + kk + 256, 0, 0);
        __builtin_prefetch(u1 + kk + 256, 0, 0);
        const v8i A0  = q2mlp_load_frag(a0 + kk);
        const v8i A1  = q2mlp_load_frag(a1 + kk);
        const v8i Bg0 = q2mlp_load_frag(g0 + kk);
        const v8i Bg1 = q2mlp_load_frag(g1 + kk);
        const v8i Bu0 = q2mlp_load_frag(u0 + kk);
        const v8i Bu1 = q2mlp_load_frag(u1 + kk);

        accg[0][0] = q2mlp_wmma(A0, Bg0, accg[0][0]);
        accg[0][1] = q2mlp_wmma(A0, Bg1, accg[0][1]);
        accg[1][0] = q2mlp_wmma(A1, Bg0, accg[1][0]);
        accg[1][1] = q2mlp_wmma(A1, Bg1, accg[1][1]);
        accu[0][0] = q2mlp_wmma(A0, Bu0, accu[0][0]);
        accu[0][1] = q2mlp_wmma(A0, Bu1, accu[0][1]);
        accu[1][0] = q2mlp_wmma(A1, Bu0, accu[1][0]);
        accu[1][1] = q2mlp_wmma(A1, Bu1, accu[1][1]);
    }

    // C/D layout: lanes 0-15 -> M = e, lanes 16-31 -> M = 8 + e; N = lane&15.
    #pragma unroll
    for (int j = 0; j < 2; ++j) {
        const int o  = o_base + j * 16 + l16;
        const float sg = wsc[o];
        const float su = wsc[o + INTER];
        #pragma unroll
        for (int i = 0; i < 2; ++i) {
            #pragma unroll
            for (int e = 0; e < 8; ++e) {
                const int m = m_base + i * 16 + half * 8 + e;
                const float sx = xs[m];
                const float g = (float)accg[i][j][e] * sx * sg;
                const float u = (float)accu[i][j][e] * sx * su;
                const float sig = 1.0f / (1.0f + __expf(-g));
                h[(size_t)m * INTER + o] = g * sig * u;
            }
        }
    }
}

// ---------------------------------------------------------------------------
// 128-point in-place FHT butterflies. H_16384 = H_128 (x) H_128, so the full
// rotation is pass 0 (stride 1 within each 128-chunk) followed by pass 1
// (stride 128 across chunks), with the 1/128 normalization folded into pass 1.
// Grid: MTOK * 128 segments, 128 threads each, 512 B LDS.
// ---------------------------------------------------------------------------
__global__ void q2mlp_fht128(float* __restrict__ data, int pass) {
    __shared__ float s[128];
    const int t = threadIdx.x;
    const unsigned seg = blockIdx.x;

    size_t base;
    int stride;
    if (pass == 0) { base = (size_t)seg * 128u;                        stride = 1;   }
    else           { base = (size_t)(seg >> 7) * INTER + (seg & 127u); stride = 128; }

    s[t] = data[base + (size_t)t * stride];
    __syncthreads();
    for (int st = 1; st < 128; st <<= 1) {
        const float mine  = s[t];
        const float other = s[t ^ st];
        __syncthreads();
        s[t] = (t & st) ? (other - mine) : (mine + other);
        __syncthreads();
    }
    float v = s[t];
    if (pass) v *= (1.0f / 128.0f);
    data[base + (size_t)t * stride] = v;
}

// ---------------------------------------------------------------------------
// GEMM2: out[m, o] = hs[m] * wds[o] * (hq[m,:] . wdq[o,:]), K = 16384.
// Wave tile 32(M) x 64(N) => 8 acc tiles, 8 WMMAs per 6 fragment loads.
// Grid: (HIDDEN/256, MTOK/64).
// ---------------------------------------------------------------------------
__global__ void q2mlp_gemm2(const signed char* __restrict__ hq,
                            const float* __restrict__ hs,
                            const signed char* __restrict__ wdq,
                            const float* __restrict__ wds,
                            float* __restrict__ out) {
    const int lane = threadIdx.x & 31;
    const int wave = threadIdx.x >> 5;
    const int wr   = wave >> 2;            // 0..1
    const int wc   = wave & 3;             // 0..3
    const int half = lane >> 4;
    const int l16  = lane & 15;

    const int m_base = blockIdx.y * 64 + wr * 32;
    const int o_base = blockIdx.x * 256 + wc * 64;

    const signed char* a0 = hq  + (size_t)(m_base + l16) * INTER + half * 8;
    const signed char* a1 = a0 + (size_t)16 * INTER;
    const signed char* b0 = wdq + (size_t)(o_base + l16) * INTER + half * 8;
    const signed char* b1 = b0 + (size_t)16 * INTER;
    const signed char* b2 = b0 + (size_t)32 * INTER;
    const signed char* b3 = b0 + (size_t)48 * INTER;

    v8i acc[2][4] = {{{}, {}, {}, {}}, {{}, {}, {}, {}}};

    for (int kk = 0; kk < INTER; kk += 64) {
        __builtin_prefetch(b0 + kk + 256, 0, 0);
        __builtin_prefetch(b1 + kk + 256, 0, 0);
        __builtin_prefetch(b2 + kk + 256, 0, 0);
        __builtin_prefetch(b3 + kk + 256, 0, 0);
        const v8i A0 = q2mlp_load_frag(a0 + kk);
        const v8i A1 = q2mlp_load_frag(a1 + kk);
        const v8i B0 = q2mlp_load_frag(b0 + kk);
        const v8i B1 = q2mlp_load_frag(b1 + kk);
        const v8i B2 = q2mlp_load_frag(b2 + kk);
        const v8i B3 = q2mlp_load_frag(b3 + kk);

        acc[0][0] = q2mlp_wmma(A0, B0, acc[0][0]);
        acc[0][1] = q2mlp_wmma(A0, B1, acc[0][1]);
        acc[0][2] = q2mlp_wmma(A0, B2, acc[0][2]);
        acc[0][3] = q2mlp_wmma(A0, B3, acc[0][3]);
        acc[1][0] = q2mlp_wmma(A1, B0, acc[1][0]);
        acc[1][1] = q2mlp_wmma(A1, B1, acc[1][1]);
        acc[1][2] = q2mlp_wmma(A1, B2, acc[1][2]);
        acc[1][3] = q2mlp_wmma(A1, B3, acc[1][3]);
    }

    #pragma unroll
    for (int j = 0; j < 4; ++j) {
        const int o  = o_base + j * 16 + l16;
        const float so = wds[o];
        #pragma unroll
        for (int i = 0; i < 2; ++i) {
            #pragma unroll
            for (int e = 0; e < 8; ++e) {
                const int m = m_base + i * 16 + half * 8 + e;
                out[(size_t)m * HIDDEN + o] = (float)acc[i][j][e] * hs[m] * so;
            }
        }
    }
}

// ---------------------------------------------------------------------------
// Launch: x -> xq; w_gate_up -> wq; w_down -> wdq; GEMM1+SiLU -> h;
// FHT x2; h -> hq; GEMM2 -> out.
// Workspace (bytes): xq 16M | wq 128M | hq 64M | wdq 64M | h(fp32) 256M |
// scales (~176 KB)  => ~528 MiB total.
// ---------------------------------------------------------------------------
extern "C" void kernel_launch(void* const* d_in, const int* in_sizes, int n_in,
                              void* d_out, int out_size, void* d_ws, size_t ws_size,
                              hipStream_t stream) {
    (void)in_sizes; (void)n_in; (void)out_size; (void)ws_size;

    const float* x   = (const float*)d_in[0];   // [MTOK, HIDDEN]
    const float* wgu = (const float*)d_in[1];   // [2*INTER, HIDDEN]
    const float* wdn = (const float*)d_in[2];   // [HIDDEN, INTER]
    float* out = (float*)d_out;                 // [MTOK, HIDDEN]

    char* ws = (char*)d_ws;
    size_t off = 0;
    signed char* xq  = (signed char*)(ws + off); off += (size_t)MTOK * HIDDEN;
    signed char* wq  = (signed char*)(ws + off); off += (size_t)(2 * INTER) * HIDDEN;
    signed char* hq  = (signed char*)(ws + off); off += (size_t)MTOK * INTER;
    signed char* wdq = (signed char*)(ws + off); off += (size_t)HIDDEN * INTER;
    float* h   = (float*)(ws + off); off += (size_t)MTOK * INTER * sizeof(float);
    float* xs  = (float*)(ws + off); off += (size_t)MTOK * sizeof(float);
    float* wsc = (float*)(ws + off); off += (size_t)(2 * INTER) * sizeof(float);
    float* hs  = (float*)(ws + off); off += (size_t)MTOK * sizeof(float);
    float* wds = (float*)(ws + off); off += (size_t)HIDDEN * sizeof(float);

    // 1) quantize activations and both weight matrices
    q2mlp_quant_rows<<<MTOK,      256, 0, stream>>>(x,   xq,  xs,  HIDDEN);
    q2mlp_quant_rows<<<2 * INTER, 256, 0, stream>>>(wgu, wq,  wsc, HIDDEN);
    q2mlp_quant_rows<<<HIDDEN,    256, 0, stream>>>(wdn, wdq, wds, INTER);

    // 2) int8 WMMA GEMM1 fused with SiLU(gate) * up  -> h
    q2mlp_gemm1_silu<<<dim3(INTER / 128, MTOK / 64), 256, 0, stream>>>(
        xq, xs, wq, wsc, h);

    // 3) Hadamard rotation: H_16384 = H_128 (x) H_128, scale 1/128 in pass 1
    q2mlp_fht128<<<MTOK * (INTER / 128), 128, 0, stream>>>(h, 0);
    q2mlp_fht128<<<MTOK * (INTER / 128), 128, 0, stream>>>(h, 1);

    // 4) quantize rotated activations
    q2mlp_quant_rows<<<MTOK, 256, 0, stream>>>(h, hq, hs, INTER);

    // 5) int8 WMMA GEMM2 -> out
    q2mlp_gemm2<<<dim3(HIDDEN / 256, MTOK / 64), 256, 0, stream>>>(
        hq, hs, wdq, wds, out);
}
